// IntraFrameNet_86363202388633
// MI455X (gfx1250) — matrix operand order repaired and checked by reading, not compiled, not run on hardware
//
#include <hip/hip_runtime.h>
#include <hip/hip_bf16.h>

// ---------------------------------------------------------------------------
// Problem constants (from reference): B=8, C=128, H=W=64 -> M=4096
// ---------------------------------------------------------------------------
#define B_DIM 8
#define C_DIM 128
#define M_DIM 4096
#define BN 32
#define NBLK (M_DIM / BN)
// s = C^(-1/4) * sqrt(log2(e)) : folded into Q/K so softmax can use exp2 directly
#define QK_SCALE 0.3571004f

typedef __attribute__((ext_vector_type(16))) _Float16 v16h;
typedef __attribute__((ext_vector_type(8)))  _Float16 v8h;
typedef __attribute__((ext_vector_type(8)))  float    v8f;
typedef __attribute__((ext_vector_type(4)))  int      v4i;

#if defined(__has_builtin)
#if __has_builtin(__builtin_amdgcn_global_load_async_to_lds_b128) && \
    __has_builtin(__builtin_amdgcn_s_wait_asynccnt)
#define USE_ASYNC_LDS 1
#endif
#endif

#if USE_ASYNC_LDS
typedef __attribute__((address_space(1))) v4i gv4;   // global 16B vector
typedef __attribute__((address_space(3))) v4i lv4;   // LDS 16B vector
#endif

// ---------------------------------------------------------------------------
// WMMA helpers (CDNA5 16x16x32 f16 -> f32)
// ---------------------------------------------------------------------------
__device__ __forceinline__ v8f wmma_f16(v16h a, v16h b, v8f c) {
  return __builtin_amdgcn_wmma_f32_16x16x32_f16(
      /*neg_a=*/false, a, /*neg_b=*/false, b,
      /*c_mod=*/(short)0, c, /*reuse_a=*/false, /*reuse_b=*/false);
}

// A fragment (16x32, MxK) from a row-major f16 matrix with leading dim `ld`.
// ISA 7.12.2: lanes 0-15 M=lane, K in {0..7,16..23}; lanes 16-31 K in {8..15,24..31}.
__device__ __forceinline__ v16h load_A_frag(const _Float16* __restrict__ p,
                                            int ld, int k0, int lane) {
  int m  = lane & 15;
  int kb = (lane & 16) ? 8 : 0;
  const _Float16* r0 = p + m * ld + k0 + kb;       // K = kb+0 .. kb+7
  const _Float16* r1 = r0 + 16;                    // K = kb+16 .. kb+23
  v16h a;
  *((v8h*)&a)       = *((const v8h*)r0);
  *(((v8h*)&a) + 1) = *((const v8h*)r1);
  return a;
}

// B fragment (32x16, KxN) where B[k][n] = src[n*ld + k0 + k] (transposed source).
// Lanes 0-15 hold N=lane, K in {0..7,16..23}; lanes 16-31 hold N=lane-16, K in {8..15,24..31}.
__device__ __forceinline__ v16h load_Bt_frag(const _Float16* __restrict__ src,
                                             int ld, int k0, int lane) {
  int n  = lane & 15;
  int kb = (lane & 16) ? 8 : 0;
  const _Float16* r0 = src + n * ld + k0 + kb;
  v16h b;
  *((v8h*)&b)       = *((const v8h*)r0);
  *(((v8h*)&b) + 1) = *((const v8h*)(r0 + 16));
  return b;
}

// Stage one 32B chunk of the K/V block: async DMA to LDS when available,
// register path otherwise. Each of 256 threads owns (srow, scs).
__device__ __forceinline__ void stage_issue(const _Float16* __restrict__ g,
                                            _Float16* __restrict__ kv,
                                            int srow, int scs) {
#if USE_ASYNC_LDS
  gv4* gp = (gv4*)g;                               // drops const, AS(1)
  lv4* lp = (lv4*)&kv[srow * C_DIM + scs];         // AS(3)
  __builtin_amdgcn_global_load_async_to_lds_b128(gp,     lp,     0, 0);
  __builtin_amdgcn_global_load_async_to_lds_b128(gp + 1, lp + 1, 0, 0);
#else
  v8h h0 = *(const v8h*)g;
  v8h h1 = *(const v8h*)(g + 8);
  *(v8h*)&kv[srow * C_DIM + scs]     = h0;
  *(v8h*)&kv[srow * C_DIM + scs + 8] = h1;
#endif
}

// ---------------------------------------------------------------------------
// Kernel 1: f32 features -> f16.
//   Qt[b][m][c]   = f[b][c][m] * QK_SCALE   (attention Q/K/V source)
//   X [b][C+c][m] = f[b][c][m]              (second half of concat input)
// ---------------------------------------------------------------------------
__global__ __launch_bounds__(256) void convert_features(
    const float* __restrict__ f, _Float16* __restrict__ Qt,
    _Float16* __restrict__ X16) {
  int idx = blockIdx.x * 256 + threadIdx.x;      // [0, B*C*M)
  int m = idx & (M_DIM - 1);
  int c = (idx >> 12) & (C_DIM - 1);
  int b = idx >> 19;
  float v = f[idx];
  Qt[((size_t)b * M_DIM + m) * C_DIM + c] = (_Float16)(v * QK_SCALE);
  X16[((size_t)b * 2 * C_DIM + C_DIM + c) * M_DIM + m] = (_Float16)v;
}

__global__ __launch_bounds__(256) void convert_w1(const float* __restrict__ w1,
                                                  _Float16* __restrict__ w1h) {
  int idx = blockIdx.x * 256 + threadIdx.x;      // [0, C*2C)
  w1h[idx] = (_Float16)w1[idx];
}

// ---------------------------------------------------------------------------
// Kernel 2: fused flash attention, transposed-S formulation with
// double-buffered async K/V staging (DMA of block i+1 overlaps compute of i).
// Grid = B * (M/128) blocks, 256 threads; wave w owns 16 query rows.
//   S^T = K * Q^T   -> per-lane softmax: VALU tree + one shfl_xor(16) each
//   O^T = V^T * P^T -> contiguous KVt A-frags, single reused P B-frag
// feats1 written transposed into X rows 0..C-1.
// ---------------------------------------------------------------------------
__global__ __launch_bounds__(256) void attn_kernel(
    const _Float16* __restrict__ Qt, _Float16* __restrict__ X16) {
  __shared__ __align__(16) _Float16 KVb [2][BN * C_DIM];   // [seq][ch]  16 KB
  __shared__ __align__(16) _Float16 KVtb[2][C_DIM * BN];   // [ch][seq]  16 KB
  __shared__ __align__(16) _Float16 Pl  [8 * 16 * 32];     // per-wave P  8 KB

  int b    = blockIdx.x >> 5;
  int mblk = blockIdx.x & 31;
  int wave = threadIdx.x >> 5;
  int lane = threadIdx.x & 31;
  int mBase = mblk * 128 + wave * 16;
  int q  = lane & 15;               // this lane's query column
  int hi = (lane & 16) ? 8 : 0;     // M-dim offset for C-layout tiles

  const _Float16* Qb = Qt + (size_t)b * M_DIM * C_DIM;

  // Q^T as B fragments (loaded once; 4 K-slices of 32)
  v16h qb[4];
#pragma unroll
  for (int kk = 0; kk < 4; ++kk)
    qb[kk] = load_Bt_frag(Qb + (size_t)mBase * C_DIM, C_DIM, 32 * kk, lane);

  v8f vzero = {};
  v8f o[8];                         // O^T: 8 channel subtiles x 16 queries
#pragma unroll
  for (int t = 0; t < 8; ++t) o[t] = vzero;
  float mi = -1e30f, li = 0.f;      // per-lane (per-query) running stats

  _Float16* Pw = Pl + wave * 512;

  int srow = threadIdx.x >> 3;            // staging: key row 0..31
  int scs  = (threadIdx.x & 7) * 16;      // staging: channel segment
  const _Float16* gstage = Qb + (size_t)srow * C_DIM + scs;

  // prologue: start DMA of block 0
  stage_issue(gstage, &KVb[0][0], srow, scs);

  for (int it = 0; it < NBLK; ++it) {
    _Float16* KV  = &KVb [it & 1][0];
    _Float16* KVt = &KVtb[it & 1][0];

#if USE_ASYNC_LDS
    __builtin_amdgcn_s_wait_asynccnt(0);     // block `it` landed in LDS
#endif
    __syncthreads();                          // everyone done with buffer it^1

    // kick off DMA of the next block into the other buffer (overlaps compute)
    if (it + 1 < NBLK)
      stage_issue(gstage + (size_t)(it + 1) * BN * C_DIM,
                  &KVb[(it + 1) & 1][0], srow, scs);

    {  // build transposed copy KVt from KV
      v8h h0 = *(const v8h*)&KV[srow * C_DIM + scs];
      v8h h1 = *(const v8h*)&KV[srow * C_DIM + scs + 8];
#pragma unroll
      for (int i = 0; i < 8; ++i) KVt[(scs + i) * BN + srow]     = h0[i];
#pragma unroll
      for (int i = 0; i < 8; ++i) KVt[(scs + 8 + i) * BN + srow] = h1[i];
    }
    __syncthreads();                          // KVt complete

    // ---- S^T = K * Q^T, software-pipelined fragment loads --------------
    v8f st0 = vzero, st1 = vzero;
    v16h a0 = load_A_frag(KV,              C_DIM, 0, lane);
    v16h a1 = load_A_frag(KV + 16 * C_DIM, C_DIM, 0, lane);
#pragma unroll
    for (int kk = 0; kk < 4; ++kk) {
      v16h na0 = a0, na1 = a1;
      if (kk < 3) {
        na0 = load_A_frag(KV,              C_DIM, 32 * (kk + 1), lane);
        na1 = load_A_frag(KV + 16 * C_DIM, C_DIM, 32 * (kk + 1), lane);
      }
      st0 = wmma_f16(a0, qb[kk], st0);
      st1 = wmma_f16(a1, qb[kk], st1);
      a0 = na0; a1 = na1;
    }

    // ---- online softmax: per-lane over 16 resident key values ----------
    float mx = fmaxf(st0[0], st1[0]);
#pragma unroll
    for (int r = 1; r < 8; ++r) mx = fmaxf(mx, fmaxf(st0[r], st1[r]));
    mx = fmaxf(mx, __shfl_xor(mx, 16, 32));   // combine key halves
    float mn    = fmaxf(mi, mx);
    float alpha = exp2f(mi - mn);
    mi = mn;

    float p0[8], p1[8], rs = 0.f;
#pragma unroll
    for (int r = 0; r < 8; ++r) {
      p0[r] = exp2f(st0[r] - mn);
      p1[r] = exp2f(st1[r] - mn);
      rs += p0[r] + p1[r];
    }
    rs += __shfl_xor(rs, 16, 32);
    li = li * alpha + rs;
#pragma unroll
    for (int t = 0; t < 8; ++t)
#pragma unroll
      for (int r = 0; r < 8; ++r) o[t][r] *= alpha;

    // ---- store P row-major [query][key]: two packed b128 LDS stores ----
    v8h h0, h1;
#pragma unroll
    for (int r = 0; r < 8; ++r) { h0[r] = (_Float16)p0[r]; h1[r] = (_Float16)p1[r]; }
    *(v8h*)&Pw[q * 32 + hi]      = h0;   // keys hi..hi+7
    *(v8h*)&Pw[q * 32 + 16 + hi] = h1;   // keys 16+hi..23+hi

    // ---- O^T += V^T * P^T, pipelined (one P fragment reused 8x) --------
    v16h pb = load_Bt_frag(Pw, 32, 0, lane);
    v16h va = load_A_frag(KVt, BN, 0, lane);
#pragma unroll
    for (int t = 0; t < 8; ++t) {
      v16h nva = va;
      if (t < 7) nva = load_A_frag(KVt + (16 * (t + 1)) * BN, BN, 0, lane);
      o[t] = wmma_f16(va, pb, o[t]);
      va = nva;
    }
  }

  // normalize (undo QK_SCALE on V, divide by softmax sum), write feats1
  // transposed into X rows 0..C-1: X[b][c][m]
  _Float16* Xb = X16 + (size_t)b * 2 * C_DIM * M_DIM;
  float inv = 1.0f / (li * QK_SCALE);
  int m = mBase + q;
#pragma unroll
  for (int t = 0; t < 8; ++t)
#pragma unroll
    for (int r = 0; r < 8; ++r) {
      int c = 16 * t + r + hi;
      Xb[(size_t)c * M_DIM + m] = (_Float16)(o[t][r] * inv);
    }
}

// ---------------------------------------------------------------------------
// Kernel 3: fused cls head.  y = w1 @ X (K=256 GEMM via WMMA), BN(inference),
// leaky_relu(0.01), then pred = w2 . y + b2 with a deterministic LDS reduction.
// Grid = B * (M/64) blocks, 256 threads. Wave w owns out-channels [16w,16w+16).
// ---------------------------------------------------------------------------
__global__ __launch_bounds__(256) void cls_head(
    const _Float16* __restrict__ X16, const _Float16* __restrict__ w1h,
    const float* __restrict__ b1, const float* __restrict__ gamma,
    const float* __restrict__ beta, const float* __restrict__ rmean,
    const float* __restrict__ rvar, const float* __restrict__ w2,
    const float* __restrict__ b2, float* __restrict__ out) {
  __shared__ __align__(16) _Float16 Xst[64 * 256];  // [m][k] transposed, 32 KB
  __shared__ float pacc[8][64];                     // per-wave partial sums

  int b  = blockIdx.x >> 6;
  int m0 = (blockIdx.x & 63) * 64;
  int t  = threadIdx.x;
  int wave = t >> 5, lane = t & 31;

  // stage X block [256 K x 64 m] transposed into LDS (thread t owns row k=t)
  const _Float16* src =
      X16 + (size_t)b * 2 * C_DIM * M_DIM + (size_t)t * M_DIM + m0;
#pragma unroll
  for (int m8 = 0; m8 < 64; m8 += 8) {
    v8h h = *(const v8h*)(src + m8);
#pragma unroll
    for (int i = 0; i < 8; ++i) Xst[(m8 + i) * 256 + t] = h[i];
  }
  __syncthreads();

  int o0  = wave * 16;
  int hi  = (lane & 16) ? 8 : 0;

  // fold conv bias + BN into scale/shift per out-channel
  float bnA[8], bnB[8], w2v[8];
#pragma unroll
  for (int r = 0; r < 8; ++r) {
    int o = o0 + r + hi;
    float sc = gamma[o] * rsqrtf(rvar[o] + 1e-5f);
    bnA[r] = sc;
    bnB[r] = (b1[o] - rmean[o]) * sc + beta[o];
    w2v[r] = w2[o];
  }

  v8f vzero = {};
  v8f acc[4];
#pragma unroll
  for (int s = 0; s < 4; ++s) acc[s] = vzero;
#pragma unroll
  for (int kk = 0; kk < 8; ++kk) {
    v16h a = load_A_frag(w1h + o0 * 2 * C_DIM, 2 * C_DIM, 32 * kk, lane);
#pragma unroll
    for (int s = 0; s < 4; ++s) {
      v16h bf = load_Bt_frag(Xst + (16 * s) * 256, 256, 32 * kk, lane);
      acc[s] = wmma_f16(a, bf, acc[s]);
    }
  }

  // BN + leaky + weighted channel reduction (deterministic)
#pragma unroll
  for (int s = 0; s < 4; ++s) {
    float partial = 0.f;
#pragma unroll
    for (int r = 0; r < 8; ++r) {
      float y = acc[s][r] * bnA[r] + bnB[r];
      y = (y >= 0.f) ? y : 0.01f * y;
      partial += w2v[r] * y;
    }
    partial += __shfl_xor(partial, 16, 32);  // combine channel halves
    if (lane < 16) pacc[wave][16 * s + lane] = partial;
  }
  __syncthreads();
  if (t < 64) {
    float sum = b2[0];
#pragma unroll
    for (int w = 0; w < 8; ++w) sum += pacc[w][t];
    out[(size_t)b * M_DIM + m0 + t] = sum;
  }
}

// ---------------------------------------------------------------------------
extern "C" void kernel_launch(void* const* d_in, const int* in_sizes, int n_in,
                              void* d_out, int out_size, void* d_ws,
                              size_t ws_size, hipStream_t stream) {
  const float* f     = (const float*)d_in[0];
  const float* w1    = (const float*)d_in[1];
  const float* b1    = (const float*)d_in[2];
  const float* gamma = (const float*)d_in[3];
  const float* beta  = (const float*)d_in[4];
  const float* rmean = (const float*)d_in[5];
  const float* rvar  = (const float*)d_in[6];
  const float* w2    = (const float*)d_in[7];
  const float* b2    = (const float*)d_in[8];
  float* out = (float*)d_out;

  char* ws = (char*)d_ws;
  _Float16* Qt  = (_Float16*)ws;                   // [B,M,C] f16 : 8 MB
  _Float16* X16 = (_Float16*)(ws + (8u << 20));    // [B,2C,M] f16: 16 MB
  _Float16* w1h = (_Float16*)(ws + (24u << 20));   // [C,2C] f16  : 64 KB

  convert_features<<<(B_DIM * C_DIM * M_DIM) / 256, 256, 0, stream>>>(f, Qt, X16);
  convert_w1<<<(C_DIM * 2 * C_DIM) / 256, 256, 0, stream>>>(w1, w1h);
  attn_kernel<<<B_DIM * (M_DIM / 128), 256, 0, stream>>>(Qt, X16);
  cls_head<<<B_DIM * (M_DIM / 64), 256, 0, stream>>>(X16, w1h, b1, gamma, beta,
                                                     rmean, rvar, w2, b2, out);
}
